// GPT_30099130811173
// MI455X (gfx1250) — matrix operand order repaired and checked by reading, not compiled, not run on hardware
//
#include <hip/hip_runtime.h>
#include <hip/hip_bf16.h>

// ---------------- model constants ----------------
constexpr int NB = 2;        // batch
constexpr int NT = 1024;     // seq len
constexpr int NC = 576;      // channels
constexpr int NH = 12;       // heads
constexpr int NL = 7;        // layers
constexpr int NV = 16640;    // vocab
constexpr int DH = 48;       // head dim (padded to 64 for WMMA)
constexpr int BT = NB * NT;  // 2048 tokens

// ---------------- WMMA / TDM types ----------------
typedef __bf16 v16bf __attribute__((ext_vector_type(16)));
typedef float  v8f   __attribute__((ext_vector_type(8)));
typedef unsigned int u32x4 __attribute__((ext_vector_type(4)));
typedef int          i32x4 __attribute__((ext_vector_type(4)));
typedef int          i32x8 __attribute__((ext_vector_type(8)));

union AFrag {
    v16bf v;
    uint4 q[2];
    unsigned short s[16];
};

__device__ inline v8f wmma_bf16(const AFrag& a, const AFrag& b, v8f c) {
    return __builtin_amdgcn_wmma_f32_16x16x32_bf16(
        /*neg_a=*/false, a.v, /*neg_b=*/false, b.v,
        /*c_mod=*/(short)0, c, /*reuse_a=*/false, /*reuse_b=*/false);
}

__device__ inline unsigned short f2bf(float f) {
    unsigned int u = __float_as_uint(f);
    u += 0x7fffu + ((u >> 16) & 1u);   // round-to-nearest-even
    return (unsigned short)(u >> 16);
}

// -------- Tensor Data Mover availability (compile-safe guard) --------
#if defined(__HIP_DEVICE_COMPILE__) && __has_builtin(__builtin_amdgcn_tensor_load_to_lds)
#define HAVE_TDM 1
#else
#define HAVE_TDM 0
#endif

#if HAVE_TDM
// Issue one 2-D TDM tile load: 64 rows x 32 bf16 columns from a row-major
// [rows, K] bf16 matrix into LDS (contiguous 64x32 tile).  D# per ISA 8.3/8.4.
__device__ inline void tdm_load_tile64x32(const unsigned short* gsrc,
                                          unsigned int lds_byte_addr,
                                          int K_, int rows_) {
    unsigned long long ga = (unsigned long long)(size_t)gsrc;
    u32x4 g0;
    g0[0] = 1u;                                            // count=1, user desc
    g0[1] = lds_byte_addr;                                 // lds_addr
    g0[2] = (unsigned int)ga;                              // global_addr[31:0]
    g0[3] = (unsigned int)((ga >> 32) & 0x01FFFFFFu)       // global_addr[56:32]
          | 0x80000000u;                                   // type=2 ("image")
    i32x8 g1;
    g1[0] = (int)(1u << 16);                               // data_size=2 bytes
    g1[1] = (int)(((unsigned)K_ & 0xFFFFu) << 16);         // tensor_dim0[15:0]
    g1[2] = (int)((((unsigned)K_ >> 16) & 0xFFFFu)         // tensor_dim0[31:16]
          | (((unsigned)rows_ & 0xFFFFu) << 16));          // tensor_dim1[15:0]
    g1[3] = (int)((((unsigned)rows_ >> 16) & 0xFFFFu)      // tensor_dim1[31:16]
          | (32u << 16));                                  // tile_dim0 = 32 elems
    g1[4] = (int)64u;                                      // tile_dim1 = 64 rows
    g1[5] = (int)(unsigned)K_;                             // dim0_stride[31:0] = K
    g1[6] = 0;                                             // stride hi / dim1_stride lo
    g1[7] = 0;
    i32x4 z4 = {0, 0, 0, 0};
#if __clang_major__ >= 23
    i32x8 z8 = {0, 0, 0, 0, 0, 0, 0, 0};
    __builtin_amdgcn_tensor_load_to_lds(g0, g1, z4, z4, z8, 0);
#else
    __builtin_amdgcn_tensor_load_to_lds(g0, g1, z4, z4, 0);
#endif
}
#endif

// ---------------- small elementwise kernels ----------------
__global__ void k_cvt_bf16(const float* __restrict__ src,
                           unsigned short* __restrict__ dst, size_t n) {
    size_t i = (size_t)blockIdx.x * blockDim.x + threadIdx.x;
    if (i < n) dst[i] = f2bf(src[i]);
}

__global__ void k_embed(const int* __restrict__ idx, const float* __restrict__ wte,
                        float* __restrict__ x, float* __restrict__ x0) {
    size_t i = (size_t)blockIdx.x * blockDim.x + threadIdx.x;
    if (i >= (size_t)BT * NC) return;
    size_t tok = i / NC, c = i % NC;
    float v = wte[(size_t)idx[tok] * NC + c];
    x[i] = v;
    x0[i] = v;
}

// mix (x = l0*x + l1*x0) + LayerNorm -> bf16 activations (statically indexed regs)
__global__ __launch_bounds__(256)
void k_mix_ln(float* __restrict__ x, const float* __restrict__ x0,
              const float* __restrict__ lambdas, int layer, int domix,
              const float* __restrict__ lw, const float* __restrict__ lb,
              unsigned short* __restrict__ hout) {
    const int tok  = blockIdx.x;
    const int tid  = threadIdx.x;
    const int lane = tid & 31, wid = tid >> 5;
    float l0 = 1.f, l1 = 0.f;
    if (domix) { l0 = lambdas[layer * 2]; l1 = lambdas[layer * 2 + 1]; }
    float* xr = x + (size_t)tok * NC;
    const float* x0r = x0 + (size_t)tok * NC;

    float vreg[3];
    float s1 = 0.f, s2 = 0.f;
#pragma unroll
    for (int it = 0; it < 3; ++it) {
        int c = tid + it * 256;
        float xm = 0.f;
        if (c < NC) {
            xm = domix ? (l0 * xr[c] + l1 * x0r[c]) : xr[c];
            if (domix) xr[c] = xm;
            s1 += xm;
            s2 += xm * xm;
        }
        vreg[it] = xm;
    }
#pragma unroll
    for (int off = 16; off; off >>= 1) {
        s1 += __shfl_xor(s1, off, 32);
        s2 += __shfl_xor(s2, off, 32);
    }
    __shared__ float sh1[8], sh2[8];
    if (lane == 0) { sh1[wid] = s1; sh2[wid] = s2; }
    __syncthreads();
    if (tid == 0) {
        float a = 0.f, bsum = 0.f;
#pragma unroll
        for (int i = 0; i < 8; ++i) { a += sh1[i]; bsum += sh2[i]; }
        sh1[0] = a; sh2[0] = bsum;
    }
    __syncthreads();
    const float mean = sh1[0] / (float)NC;
    const float var  = sh2[0] / (float)NC - mean * mean;
    const float rstd = rsqrtf(var + 1e-5f);
#pragma unroll
    for (int it = 0; it < 3; ++it) {
        int c = tid + it * 256;
        if (c < NC)
            hout[(size_t)tok * NC + c] = f2bf((vreg[it] - mean) * rstd * lw[c] + lb[c]);
    }
}

// split QKV -> padded per-head bf16 [B*H, T, 64]; scale folded into q
__global__ void k_split_qkv(const float* __restrict__ qkv,
                            unsigned short* __restrict__ qp,
                            unsigned short* __restrict__ kp,
                            unsigned short* __restrict__ vp) {
    size_t i = (size_t)blockIdx.x * blockDim.x + threadIdx.x;
    if (i >= (size_t)BT * NH * 64) return;
    int d = (int)(i & 63);
    size_t rest = i >> 6;
    int hh = (int)(rest % NH);
    size_t tok = rest / NH;
    int b = (int)(tok / NT), ti = (int)(tok % NT);
    size_t dst = ((size_t)(b * NH + hh) * NT + ti) * 64 + d;
    float qv = 0.f, kv = 0.f, vv = 0.f;
    if (d < DH) {
        size_t base = tok * (size_t)(3 * NC) + (size_t)hh * DH + d;
        qv = qkv[base] * 0.1443375673f; // 1/sqrt(48)
        kv = qkv[base + NC];
        vv = qkv[base + 2 * NC];
    }
    qp[dst] = f2bf(qv);
    kp[dst] = f2bf(kv);
    vp[dst] = f2bf(vv);
}

__global__ void k_geglu(const float* __restrict__ gate, const float* __restrict__ up,
                        unsigned short* __restrict__ out) {
    size_t i = (size_t)blockIdx.x * blockDim.x + threadIdx.x;
    if (i >= (size_t)BT * 2 * NC) return;
    float g = gate[i];
    float t = 0.7978845608f * (g + 0.044715f * g * g * g);
    float gl = 0.5f * g * (1.f + tanhf(t));
    out[i] = f2bf(gl * up[i]);
}

// ---------------- tiled WMMA GEMM: out[M,N] = A[M,K](bf16) @ W[N,K]^T ----------------
// EPI: 0 = store f32, 1 = +residual, 2 = 30*tanh(x/30) (logit clamp)
// Requires M%64==0, N%64==0, K%32==0 (true for all shapes here).
// TDM path: wave 0 DMAs double-buffered 64x32 A/B tiles into LDS via
// tensor_load_to_lds; compute waves consume behind s_wait_tensorcnt + barrier.
template <int EPI>
__global__ __launch_bounds__(256)
void k_gemm(float* __restrict__ out,
            const unsigned short* __restrict__ A,
            const unsigned short* __restrict__ W,
            const float* __restrict__ res, int M, int N, int K) {
    const int tid = threadIdx.x;
    const int lane = tid & 31, w = tid >> 5;
    const int wm = w >> 1, wn = w & 1;       // 4x2 wave grid
    const int m0 = blockIdx.y * 64, n0 = blockIdx.x * 64;
    const int ml = lane & 15, half = lane >> 4;

    v8f acc0 = {}, acc1 = {};

#if HAVE_TDM
    __shared__ __align__(16) unsigned short As[2][64 * 32];
    __shared__ __align__(16) unsigned short Bs[2][64 * 32];
    const unsigned int asOff0 = (unsigned int)(size_t)(&As[0][0]);
    const unsigned int asOff1 = (unsigned int)(size_t)(&As[1][0]);
    const unsigned int bsOff0 = (unsigned int)(size_t)(&Bs[0][0]);
    const unsigned int bsOff1 = (unsigned int)(size_t)(&Bs[1][0]);
    int cur = 0;
    if (tid < 32) {
        tdm_load_tile64x32(A + (size_t)m0 * K, asOff0, K, M);
        tdm_load_tile64x32(W + (size_t)n0 * K, bsOff0, K, N);
    }
    for (int k0 = 0; k0 < K; k0 += 32) {
        const bool more = (k0 + 32) < K;
        if (tid < 32) {
            if (more) {
                tdm_load_tile64x32(A + (size_t)m0 * K + k0 + 32,
                                   cur ? asOff0 : asOff1, K, M);
                tdm_load_tile64x32(W + (size_t)n0 * K + k0 + 32,
                                   cur ? bsOff0 : bsOff1, K, N);
                __builtin_amdgcn_s_wait_tensorcnt(2);
            } else {
                __builtin_amdgcn_s_wait_tensorcnt(0);
            }
        }
        __syncthreads();
        const unsigned short* Asb = &As[cur][0];
        const unsigned short* Bsb = &Bs[cur][0];
#else
    __shared__ __align__(16) unsigned short AsBuf[64 * 32];
    __shared__ __align__(16) unsigned short BsBuf[64 * 32];
    const int row = tid >> 2, col8 = (tid & 3) * 8;   // cooperative loader coords
    for (int k0 = 0; k0 < K; k0 += 32) {
        __syncthreads();
        *(uint4*)(&AsBuf[row * 32 + col8]) =
            *(const uint4*)(&A[(size_t)(m0 + row) * K + k0 + col8]);
        *(uint4*)(&BsBuf[row * 32 + col8]) =
            *(const uint4*)(&W[(size_t)(n0 + row) * K + k0 + col8]);
        if (k0 + 32 < K) {
            __builtin_prefetch(&A[(size_t)(m0 + row) * K + k0 + 32 + col8], 0, 3);
            __builtin_prefetch(&W[(size_t)(n0 + row) * K + k0 + 32 + col8], 0, 3);
        }
        __syncthreads();
        const unsigned short* Asb = &AsBuf[0];
        const unsigned short* Bsb = &BsBuf[0];
#endif
        AFrag a, b0, b1;
        const unsigned short* ar = &Asb[(wm * 16 + ml) * 32];
        a.q[0] = *(const uint4*)(ar + half * 16);
        a.q[1] = *(const uint4*)(ar + half * 16 + 8);
        const unsigned short* b0r = &Bsb[(wn * 32 + ml) * 32 + half * 16];
        b0.q[0] = *(const uint4*)(b0r);
        b0.q[1] = *(const uint4*)(b0r + 8);
        const unsigned short* b1r = &Bsb[(wn * 32 + 16 + ml) * 32 + half * 16];
        b1.q[0] = *(const uint4*)(b1r);
        b1.q[1] = *(const uint4*)(b1r + 8);

        acc0 = wmma_bf16(a, b0, acc0);
        acc1 = wmma_bf16(a, b1, acc1);
#if HAVE_TDM
        __syncthreads();
        cur ^= 1;
    }
#else
    }
#endif

#pragma unroll
    for (int r = 0; r < 8; ++r) {
        int grow = m0 + wm * 16 + r + 8 * half;
        int gc0  = n0 + wn * 32 + ml;
        size_t o0 = (size_t)grow * N + gc0;
        size_t o1 = o0 + 16;
        float v0 = acc0[r], v1 = acc1[r];
        if (EPI == 1) { v0 += res[o0]; v1 += res[o1]; }
        if (EPI == 2) {
            v0 = 30.f * tanhf(v0 * (1.f / 30.f));
            v1 = 30.f * tanhf(v1 * (1.f / 30.f));
        }
        out[o0] = v0;
        out[o1] = v1;
    }
}

// ---------------- flash attention, one wave per 16-query tile ----------------
__global__ __launch_bounds__(32)
void k_attn(const unsigned short* __restrict__ qpad,
            const unsigned short* __restrict__ kpad,
            const unsigned short* __restrict__ vpad,
            unsigned short* __restrict__ ybf) {
    __shared__ __align__(16) unsigned short Pt[16 * 32];
    __shared__ __align__(16) unsigned short Vt[64 * 32];
    const int qt = blockIdx.x, bh = blockIdx.y;
    const int b = bh / NH, h = bh % NH;
    const int lane = threadIdx.x;
    const int m = lane & 15, half = lane >> 4;
    const float slope = exp2f(-8.0f * (float)(h + 1) / (float)NH);

    const unsigned short* Qb = qpad + (size_t)bh * NT * 64;
    const unsigned short* Kb = kpad + (size_t)bh * NT * 64;
    const unsigned short* Vb = vpad + (size_t)bh * NT * 64;

    AFrag aq0, aq1;
    {
        const unsigned short* qr = Qb + (size_t)(qt * 16 + m) * 64;
        aq0.q[0] = *(const uint4*)(qr + half * 16);
        aq0.q[1] = *(const uint4*)(qr + half * 16 + 8);
        aq1.q[0] = *(const uint4*)(qr + 32 + half * 16);
        aq1.q[1] = *(const uint4*)(qr + 32 + half * 16 + 8);
    }

    v8f zero = {};
    v8f oacc0 = zero, oacc1 = zero, oacc2 = zero, oacc3 = zero;
    float mrow[8], lrow[8];
#pragma unroll
    for (int r = 0; r < 8; ++r) { mrow[r] = -1e30f; lrow[r] = 0.f; }

    const int qlast = qt * 16 + 15;
    for (int key0 = 0; key0 <= qlast; key0 += 32) {
        // stage V transposed: Vt[d][key]  (vectorized global reads, scalar LDS scatter)
        {
            const uint4* vr4 = (const uint4*)(Vb + (size_t)(key0 + lane) * 64);
#pragma unroll
            for (int blk = 0; blk < 8; ++blk) {
                uint4 t = vr4[blk];
                const unsigned short* ts = (const unsigned short*)&t;
#pragma unroll
                for (int jj = 0; jj < 8; ++jj)
                    Vt[(blk * 8 + jj) * 32 + lane] = ts[jj];
            }
        }
        // S = Q K^T over padded d=64 (scale already folded into Q)
        v8f s0, s1;
        {
            AFrag bk0, bk1;
            const unsigned short* kr = Kb + (size_t)(key0 + m) * 64;
            bk0.q[0] = *(const uint4*)(kr + half * 16);
            bk0.q[1] = *(const uint4*)(kr + half * 16 + 8);
            bk1.q[0] = *(const uint4*)(kr + 32 + half * 16);
            bk1.q[1] = *(const uint4*)(kr + 32 + half * 16 + 8);
            v8f z = zero;
            z  = wmma_bf16(aq0, bk0, z);
            s0 = wmma_bf16(aq1, bk1, z);
            const unsigned short* kr2 = Kb + (size_t)(key0 + 16 + m) * 64;
            bk0.q[0] = *(const uint4*)(kr2 + half * 16);
            bk0.q[1] = *(const uint4*)(kr2 + half * 16 + 8);
            bk1.q[0] = *(const uint4*)(kr2 + 32 + half * 16);
            bk1.q[1] = *(const uint4*)(kr2 + 32 + half * 16 + 8);
            z  = wmma_bf16(aq0, bk0, zero);
            s1 = wmma_bf16(aq1, bk1, z);
        }
        // online softmax with ALiBi + causal mask (fully unrolled: no VGPR indexing)
#pragma unroll
        for (int r = 0; r < 8; ++r) {
            const int qrow = qt * 16 + r + 8 * half;
            const int kA = key0 + m, kB = key0 + 16 + m;
            float a0 = s0[r] + slope * (kA > qrow ? (float)(kA - qrow) : 0.f);
            float a1 = s1[r] + slope * (kB > qrow ? (float)(kB - qrow) : 0.f);
            if (kA > qrow) a0 = -1e30f;
            if (kB > qrow) a1 = -1e30f;
            float mx = fmaxf(a0, a1);
#pragma unroll
            for (int off = 1; off < 16; off <<= 1) mx = fmaxf(mx, __shfl_xor(mx, off, 16));
            float mnew = fmaxf(mrow[r], mx);
            float corr = __expf(mrow[r] - mnew);
            float p0 = __expf(a0 - mnew);
            float p1 = __expf(a1 - mnew);
            float ps = p0 + p1;
#pragma unroll
            for (int off = 1; off < 16; off <<= 1) ps += __shfl_xor(ps, off, 16);
            mrow[r] = mnew;
            lrow[r] = lrow[r] * corr + ps;
            oacc0[r] *= corr; oacc1[r] *= corr;
            oacc2[r] *= corr; oacc3[r] *= corr;
            Pt[(r + 8 * half) * 32 + m]      = f2bf(p0);
            Pt[(r + 8 * half) * 32 + 16 + m] = f2bf(p1);
        }
        // P (transposed through LDS) @ V
        AFrag ap;
        const unsigned short* pr = &Pt[m * 32];
        ap.q[0] = *(const uint4*)(pr + half * 16);
        ap.q[1] = *(const uint4*)(pr + half * 16 + 8);
        AFrag bv;
        const unsigned short* vt0 = &Vt[(0 * 16 + m) * 32 + half * 16];
        bv.q[0] = *(const uint4*)(vt0); bv.q[1] = *(const uint4*)(vt0 + 8);
        oacc0 = wmma_bf16(ap, bv, oacc0);
        const unsigned short* vt1 = &Vt[(1 * 16 + m) * 32 + half * 16];
        bv.q[0] = *(const uint4*)(vt1); bv.q[1] = *(const uint4*)(vt1 + 8);
        oacc1 = wmma_bf16(ap, bv, oacc1);
        const unsigned short* vt2 = &Vt[(2 * 16 + m) * 32 + half * 16];
        bv.q[0] = *(const uint4*)(vt2); bv.q[1] = *(const uint4*)(vt2 + 8);
        oacc2 = wmma_bf16(ap, bv, oacc2);
        const unsigned short* vt3 = &Vt[(3 * 16 + m) * 32 + half * 16];
        bv.q[0] = *(const uint4*)(vt3); bv.q[1] = *(const uint4*)(vt3 + 8);
        oacc3 = wmma_bf16(ap, bv, oacc3);
    }
    // normalize and scatter (drop padded d>=48; oacc3 is padding)
#pragma unroll
    for (int r = 0; r < 8; ++r) {
        float inv = 1.0f / lrow[r];
        int tok = b * NT + qt * 16 + r + 8 * half;
        ybf[(size_t)tok * NC + h * DH + 0  + m] = f2bf(oacc0[r] * inv);
        ybf[(size_t)tok * NC + h * DH + 16 + m] = f2bf(oacc1[r] * inv);
        ybf[(size_t)tok * NC + h * DH + 32 + m] = f2bf(oacc2[r] * inv);
    }
}

// ---------------- loss ----------------
__global__ __launch_bounds__(256)
void k_nll(const float* __restrict__ logits, const int* __restrict__ targets,
           float* __restrict__ nll) {
    const int tok = blockIdx.x, tid = threadIdx.x;
    const int lane = tid & 31, wid = tid >> 5;
    const float* row = logits + (size_t)tok * NV;
    __shared__ float sh[8];
    float mx = -1e30f;
    for (int c = tid; c < NV; c += 256) mx = fmaxf(mx, row[c]);
#pragma unroll
    for (int off = 16; off; off >>= 1) mx = fmaxf(mx, __shfl_xor(mx, off, 32));
    if (lane == 0) sh[wid] = mx;
    __syncthreads();
    if (tid == 0) {
        float t = sh[0];
#pragma unroll
        for (int i = 1; i < 8; ++i) t = fmaxf(t, sh[i]);
        sh[0] = t;
    }
    __syncthreads();
    mx = sh[0];
    __syncthreads();
    float s = 0.f;
    for (int c = tid; c < NV; c += 256) s += __expf(row[c] - mx);
#pragma unroll
    for (int off = 16; off; off >>= 1) s += __shfl_xor(s, off, 32);
    if (lane == 0) sh[wid] = s;
    __syncthreads();
    if (tid == 0) {
        float t = 0.f;
#pragma unroll
        for (int i = 0; i < 8; ++i) t += sh[i];
        float lp = row[targets[tok]] - mx - logf(t);
        nll[tok] = -lp;
    }
}

__global__ __launch_bounds__(256)
void k_loss_reduce(const float* __restrict__ nll, float* __restrict__ loss) {
    const int tid = threadIdx.x;
    __shared__ float sh[256];
    float s = 0.f;
    for (int i = tid; i < BT; i += 256) s += nll[i];
    sh[tid] = s;
    __syncthreads();
    for (int st = 128; st; st >>= 1) {
        if (tid < st) sh[tid] += sh[tid + st];
        __syncthreads();
    }
    if (tid == 0) *loss = sh[0] / (float)BT;
}

// ---------------- host side ----------------
static void launch_gemm(int epi, float* out, const unsigned short* A,
                        const unsigned short* W, const float* res,
                        int M, int N, int K, hipStream_t s) {
    dim3 grid(N / 64, M / 64), block(256);
    if (epi == 0)      k_gemm<0><<<grid, block, 0, s>>>(out, A, W, res, M, N, K);
    else if (epi == 1) k_gemm<1><<<grid, block, 0, s>>>(out, A, W, res, M, N, K);
    else               k_gemm<2><<<grid, block, 0, s>>>(out, A, W, res, M, N, K);
}

extern "C" void kernel_launch(void* const* d_in, const int* in_sizes, int n_in,
                              void* d_out, int out_size, void* d_ws, size_t ws_size,
                              hipStream_t stream) {
    const int*   idx     = (const int*)d_in[0];
    const int*   targets = (const int*)d_in[1];
    const float* wte     = (const float*)d_in[2];
    const float* lambdas = (const float*)d_in[3];
    const float* ln1w    = (const float*)d_in[4];
    const float* ln1b    = (const float*)d_in[5];
    const float* ln2w    = (const float*)d_in[6];
    const float* ln2b    = (const float*)d_in[7];
    const float* wqkv    = (const float*)d_in[8];
    const float* wproj   = (const float*)d_in[9];
    const float* wgate   = (const float*)d_in[10];
    const float* wup     = (const float*)d_in[11];
    const float* wdown   = (const float*)d_in[12];
    const float* lnfw    = (const float*)d_in[13];
    const float* lnfb    = (const float*)d_in[14];

    // workspace bump allocator (~128 MB total)
    char* p = (char*)d_ws;
    auto alloc = [&](size_t bytes) -> char* {
        char* r = p;
        p += (bytes + 255) & ~(size_t)255;
        return r;
    };
    unsigned short* wte_bf   = (unsigned short*)alloc((size_t)NV * NC * 2);
    unsigned short* wqkv_bf  = (unsigned short*)alloc((size_t)NL * 3 * NC * NC * 2);
    unsigned short* wproj_bf = (unsigned short*)alloc((size_t)NL * NC * NC * 2);
    unsigned short* wgate_bf = (unsigned short*)alloc((size_t)NL * 2 * NC * NC * 2);
    unsigned short* wup_bf   = (unsigned short*)alloc((size_t)NL * 2 * NC * NC * 2);
    unsigned short* wdown_bf = (unsigned short*)alloc((size_t)NL * NC * 2 * NC * 2);
    float*          x        = (float*)alloc((size_t)BT * NC * 4);
    float*          x0       = (float*)alloc((size_t)BT * NC * 4);
    unsigned short* hbf      = (unsigned short*)alloc((size_t)BT * NC * 2);
    float*          qkv      = (float*)alloc((size_t)BT * 3 * NC * 4);
    unsigned short* qpad     = (unsigned short*)alloc((size_t)NB * NH * NT * 64 * 2);
    unsigned short* kpad     = (unsigned short*)alloc((size_t)NB * NH * NT * 64 * 2);
    unsigned short* vpad     = (unsigned short*)alloc((size_t)NB * NH * NT * 64 * 2);
    unsigned short* ybf      = (unsigned short*)alloc((size_t)BT * NC * 2);
    float*          gatebuf  = (float*)alloc((size_t)BT * 2 * NC * 4);
    float*          upbuf    = (float*)alloc((size_t)BT * 2 * NC * 4);
    unsigned short* gubf     = (unsigned short*)alloc((size_t)BT * 2 * NC * 2);
    float*          nllbuf   = (float*)alloc((size_t)BT * 4);

    auto cvt = [&](const float* src, unsigned short* dst, size_t n) {
        k_cvt_bf16<<<dim3((unsigned)((n + 255) / 256)), dim3(256), 0, stream>>>(src, dst, n);
    };
    cvt(wte,   wte_bf,   (size_t)NV * NC);
    cvt(wqkv,  wqkv_bf,  (size_t)NL * 3 * NC * NC);
    cvt(wproj, wproj_bf, (size_t)NL * NC * NC);
    cvt(wgate, wgate_bf, (size_t)NL * 2 * NC * NC);
    cvt(wup,   wup_bf,   (size_t)NL * 2 * NC * NC);
    cvt(wdown, wdown_bf, (size_t)NL * NC * 2 * NC);

    {
        size_t n = (size_t)BT * NC;
        k_embed<<<dim3((unsigned)((n + 255) / 256)), dim3(256), 0, stream>>>(idx, wte, x, x0);
    }

    for (int l = 0; l < NL; ++l) {
        k_mix_ln<<<dim3(BT), dim3(256), 0, stream>>>(
            x, x0, lambdas, l, 1, ln1w + (size_t)l * NC, ln1b + (size_t)l * NC, hbf);
        launch_gemm(0, qkv, hbf, wqkv_bf + (size_t)l * 3 * NC * NC, nullptr,
                    BT, 3 * NC, NC, stream);
        {
            size_t n = (size_t)BT * NH * 64;
            k_split_qkv<<<dim3((unsigned)((n + 255) / 256)), dim3(256), 0, stream>>>(
                qkv, qpad, kpad, vpad);
        }
        {
            dim3 g(NT / 16, NB * NH);
            k_attn<<<g, dim3(32), 0, stream>>>(qpad, kpad, vpad, ybf);
        }
        launch_gemm(1, x, ybf, wproj_bf + (size_t)l * NC * NC, x, BT, NC, NC, stream);
        k_mix_ln<<<dim3(BT), dim3(256), 0, stream>>>(
            x, x0, lambdas, l, 0, ln2w + (size_t)l * NC, ln2b + (size_t)l * NC, hbf);
        launch_gemm(0, gatebuf, hbf, wgate_bf + (size_t)l * 2 * NC * NC, nullptr,
                    BT, 2 * NC, NC, stream);
        launch_gemm(0, upbuf, hbf, wup_bf + (size_t)l * 2 * NC * NC, nullptr,
                    BT, 2 * NC, NC, stream);
        {
            size_t n = (size_t)BT * 2 * NC;
            k_geglu<<<dim3((unsigned)((n + 255) / 256)), dim3(256), 0, stream>>>(
                gatebuf, upbuf, gubf);
        }
        launch_gemm(1, x, gubf, wdown_bf + (size_t)l * NC * 2 * NC, x,
                    BT, NC, 2 * NC, stream);
    }

    // final LN + tied LM head with logit clamp (writes d_out logits)
    k_mix_ln<<<dim3(BT), dim3(256), 0, stream>>>(x, x0, lambdas, 0, 0, lnfw, lnfb, hbf);
    float* logits = (float*)d_out;
    launch_gemm(2, logits, hbf, wte_bf, nullptr, BT, NV, NC, stream);

    // loss
    k_nll<<<dim3(BT), dim3(256), 0, stream>>>(logits, targets, nllbuf);
    k_loss_reduce<<<dim3(1), dim3(256), 0, stream>>>(nllbuf, logits + (size_t)BT * NV);
}
